// SparseConvNet_Triplane_48249662603663
// MI455X (gfx1250) — matrix operand order repaired and checked by reading, not compiled, not run on hardware
//
#include <hip/hip_runtime.h>
#include <hip/hip_bf16.h>

typedef __attribute__((ext_vector_type(16))) _Float16 v16h;
typedef __attribute__((ext_vector_type(8)))  float    v8f;

union FragH { uint4 u[2]; v16h h; };
union Out8  { uint4 u; _Float16 h[8]; };

// ---------------------------------------------------------------- utilities

__global__ void k_zero16(uint4* __restrict__ p, size_t n16) {
  size_t i = (size_t)blockIdx.x * blockDim.x + threadIdx.x;
  size_t stride = (size_t)gridDim.x * blockDim.x;
  uint4 z; z.x = z.y = z.z = z.w = 0u;
  for (; i < n16; i += stride) p[i] = z;
}

__global__ void k_mask_to_float(const unsigned char* __restrict__ mb,
                                float* __restrict__ m, int n) {
  int t = blockIdx.x * 256 + threadIdx.x;
  if (t < n) m[t] = mb[t] ? 1.0f : 0.0f;
}

// maxpool 3x3x3 stride 2 pad 1 (init 0) on a 0/1 float field
__global__ void k_downsample_mask(const float* __restrict__ src,
                                  float* __restrict__ dst, int Do, int Di) {
  int t = blockIdx.x * 256 + threadIdx.x;
  if (t >= Do * Do * Do) return;
  int x = t % Do, y = (t / Do) % Do, z = t / (Do * Do);
  float m = 0.0f;
  for (int dz = -1; dz <= 1; ++dz)
    for (int dy = -1; dy <= 1; ++dy)
      for (int dx = -1; dx <= 1; ++dx) {
        int zz = 2 * z + dz, yy = 2 * y + dy, xx = 2 * x + dx;
        if (zz >= 0 && zz < Di && yy >= 0 && yy < Di && xx >= 0 && xx < Di)
          m = fmaxf(m, src[(zz * Di + yy) * Di + xx]);
      }
  dst[t] = m;
}

// x (NCDHW f32) * mask -> padded 66^3 x 32ch channels-last f16 (ch 16..31 stay zero)
__global__ void k_embed_input(const float* __restrict__ x,
                              const float* __restrict__ m0,
                              _Float16* __restrict__ dst) {
  int t = blockIdx.x * 256 + threadIdx.x;   // 16 * 64^3 threads
  if (t >= (16 << 18)) return;
  int c = t & 15;
  int v = t >> 4;
  int xx = v & 63, yy = (v >> 6) & 63, zz = v >> 12;
  float val = x[(size_t)c * 262144 + v] * m0[v];
  dst[((((size_t)(zz + 1)) * 66 + (yy + 1)) * 66 + (xx + 1)) * 32 + c] = (_Float16)val;
}

// bn (4,C): gamma,beta,mu,var -> scale = g*rsqrt(v+eps), shift = b - mu*scale
__global__ void k_bn_prep(const float* __restrict__ bn, float* __restrict__ scale,
                          float* __restrict__ shift, int C) {
  int c = blockIdx.x * 64 + threadIdx.x;
  if (c >= C) return;
  float g = bn[c], b = bn[C + c], mu = bn[2 * C + c], v = bn[3 * C + c];
  float s = g * rsqrtf(v + 1e-3f);
  scale[c] = s;
  shift[c] = b - mu * s;
}

// Pack W[co][srcCin][3][3][3] f32 -> A-fragment order f16, K = tap*CinPad + ci.
// A order used here (contiguous k, B reads the same) per step:
//   lane<16: row M=lane,    K offsets {0..15}
//   lane>=16: row M=lane-16, K offsets {16..31}
// ci >= srcCin zero-padded (input channel padding for the first layer).
__global__ void k_pack_weights(const float* __restrict__ w, _Float16* __restrict__ dst,
                               int CinPad, int srcCin, int Cout, int nSteps) {
  int t = blockIdx.x * 256 + threadIdx.x;
  int total = 32 * Cout * nSteps;           // (Cout/16)*nSteps*32*16
  if (t >= total) return;
  int e      = t & 15;
  int lane   = (t >> 4) & 31;
  int q      = t >> 9;
  int kStep  = q % nSteps;
  int coTile = q / nSteps;
  int co = coTile * 16 + (lane & 15);
  int h  = (lane & 16);                     // 0 or 16
  int j  = e;
  int kk = kStep * 32 + h + j;
  float val = 0.0f;
  if (kk < 27 * CinPad) {
    int tap = kk / CinPad;
    int ci  = kk - tap * CinPad;
    if (ci < srcCin) val = w[((size_t)co * srcCin + ci) * 27 + tap];
  }
  dst[t] = (_Float16)val;
}

// ------------------------------------------------- implicit-GEMM conv + BN/ReLU/mask
// One wave per (32 Cout) x (16 voxel) tile: two 16x16 accumulators share each B
// fragment. K loop fully unrolled; all fragment loads are base+immediate.
template <int CIN, int PIN>
__launch_bounds__(32)
__global__ void k_conv3d_wmma(const _Float16* __restrict__ in,
                              const _Float16* __restrict__ Apack,
                              const float* __restrict__ scale,
                              const float* __restrict__ shift,
                              const float* __restrict__ mask,
                              _Float16* __restrict__ out, int Pout, int Cout,
                              int Dout, int dsh, int stride) {
  constexpr int NSTEPS = 27 * CIN / 32;
  const int lane = threadIdx.x;
  const int n  = blockIdx.x * 16 + (lane & 15);     // linear output voxel
  const int xo = n & (Dout - 1);
  const int yo = (n >> dsh) & (Dout - 1);
  const int zo = n >> (2 * dsh);
  // per-lane B base: output voxel's (0,0,0) tap, plus the K-half select
  const size_t vbase =
      ((size_t)((zo * stride) * PIN + (yo * stride)) * PIN + (xo * stride)) * CIN;
  const _Float16* bp = in + vbase + (lane & 16);

  const int coT0 = blockIdx.y * 2;                  // two 16-wide Cout tiles
  const _Float16* ap0 = Apack + ((size_t)(coT0 * NSTEPS) * 32 + lane) * 16;
  const _Float16* ap1 = ap0 + (size_t)NSTEPS * 512;

  v8f acc0 = {}, acc1 = {};
  int s = 0;
#pragma unroll
  for (int tap = 0; tap < 27; ++tap) {
    const int dz = tap / 9, dyy = (tap / 3) % 3, dxx = tap % 3;
    const size_t toff = ((size_t)(dz * PIN + dyy) * PIN + dxx) * CIN;  // constant
#pragma unroll
    for (int c = 0; c < CIN; c += 32, ++s) {
      FragH b;
      const uint4* bv = (const uint4*)(bp + toff + c);   // imm-offset b128 pair
      b.u[0] = bv[0];
      b.u[1] = bv[1];
      FragH a0, a1;
      const uint4* av0 = (const uint4*)(ap0 + (size_t)s * 512);
      const uint4* av1 = (const uint4*)(ap1 + (size_t)s * 512);
      a0.u[0] = av0[0];
      a0.u[1] = av0[1];
      a1.u[0] = av1[0];
      a1.u[1] = av1[1];
      acc0 = __builtin_amdgcn_wmma_f32_16x16x32_f16(
          false, a0.h, false, b.h, (short)0, acc0, false, false);
      acc1 = __builtin_amdgcn_wmma_f32_16x16x32_f16(
          false, a1.h, false, b.h, (short)0, acc1, false, false);
    }
  }

  // C/D layout: lane holds N=lane&15, rows M = r + (lane<16?0:8)
  const float mv = mask[n];
  const int coBase0 = coT0 * 16 + ((lane & 16) >> 1);
  const int coBase1 = coBase0 + 16;
  Out8 o0, o1;
#pragma unroll
  for (int r = 0; r < 8; ++r) {
    float y0 = fmaf(acc0[r], scale[coBase0 + r], shift[coBase0 + r]);
    o0.h[r] = (_Float16)(fmaxf(y0, 0.0f) * mv);
    float y1 = fmaf(acc1[r], scale[coBase1 + r], shift[coBase1 + r]);
    o1.h[r] = (_Float16)(fmaxf(y1, 0.0f) * mv);
  }
  size_t voxo = (size_t)(((zo + 1) * Pout + (yo + 1)) * Pout + (xo + 1)) * Cout;
  *(uint4*)(out + voxo + coBase0) = o0.u;
  *(uint4*)(out + voxo + coBase1) = o1.u;
}

// ---------------------------------------------------------------- triplane + sample

__global__ void k_make_planes(const _Float16* __restrict__ act, float* __restrict__ pxy,
                              float* __restrict__ pyz, float* __restrict__ pxz) {
  int t = blockIdx.x * 256 + threadIdx.x;   // 128*16*16
  if (t >= 32768) return;
  int j = t & 15, i = (t >> 4) & 15, c = t >> 8;
  float sxy = 0.f, syz = 0.f, sxz = 0.f;
  for (int q = 0; q < 16; ++q) {
    sxy += (float)act[((((size_t)(q + 1)) * 18 + (i + 1)) * 18 + (j + 1)) * 128 + c]; // mean z
    syz += (float)act[((((size_t)(i + 1)) * 18 + (j + 1)) * 18 + (q + 1)) * 128 + c]; // mean x
    sxz += (float)act[((((size_t)(i + 1)) * 18 + (q + 1)) * 18 + (j + 1)) * 128 + c]; // mean y
  }
  int o = c * 256 + i * 16 + j;
  pxy[o] = sxy * 0.0625f;   // [c][y][x]
  pyz[o] = syz * 0.0625f;   // [c][z][y]
  pxz[o] = sxz * 0.0625f;   // [c][z][x]
}

__device__ __forceinline__ float bilin16(const float* __restrict__ img, float gx, float gy) {
  float ix = (gx + 1.0f) * 0.5f * 15.0f;
  float iy = (gy + 1.0f) * 0.5f * 15.0f;
  float x0f = floorf(ix), y0f = floorf(iy);
  float wx = ix - x0f, wy = iy - y0f;
  int x0 = (int)x0f, y0 = (int)y0f;
  float v = 0.0f;
#pragma unroll
  for (int dy = 0; dy < 2; ++dy)
#pragma unroll
    for (int dx = 0; dx < 2; ++dx) {
      int xx = x0 + dx, yy = y0 + dy;
      float w = (dy ? wy : 1.0f - wy) * (dx ? wx : 1.0f - wx);
      if (xx >= 0 && xx < 16 && yy >= 0 && yy < 16) v += img[yy * 16 + xx] * w;
    }
  return v;
}

__launch_bounds__(128)
__global__ void k_sample(const float* __restrict__ coords, const float* __restrict__ pxy,
                         const float* __restrict__ pyz, const float* __restrict__ pxz,
                         float* __restrict__ out) {
  int p = blockIdx.x;
  int c = threadIdx.x;
  float c0 = coords[p * 3 + 0], c1 = coords[p * 3 + 1], c2 = coords[p * 3 + 2];
  float r = bilin16(pxy + c * 256, c0, c1)
          + bilin16(pyz + c * 256, c1, c2)
          + bilin16(pxz + c * 256, c0, c2);
  out[(size_t)p * 128 + c] = r;
}

// ---------------------------------------------------------------- host launcher

static inline int lg2i(int v) { int r = 0; while ((1 << r) < v) ++r; return r; }

extern "C" void kernel_launch(void* const* d_in, const int* in_sizes, int n_in,
                              void* d_out, int out_size, void* d_ws, size_t ws_size,
                              hipStream_t stream) {
  (void)in_sizes; (void)n_in; (void)out_size;
  const float*         xin   = (const float*)d_in[0];
  const unsigned char* maskb = (const unsigned char*)d_in[1];
  const float*         crd   = (const float*)d_in[2];

  char* ws = (char*)d_ws;
  size_t off = 0;
  auto alloc = [&](size_t bytes) -> size_t {
    size_t o = off;
    off = (off + bytes + 255) & ~(size_t)255;
    return o;
  };

  const size_t V0 = 66ull * 66 * 66, V1 = 34ull * 34 * 34, V2 = 18ull * 18 * 18;
  // f16 activation tensors (padded, channels-last) -- contiguous so one zero pass
  size_t oTin = alloc(V0 * 32 * 2);   // 16 real + 16 zero-pad channels
  size_t oT0a = alloc(V0 * 32 * 2);
  size_t oT0b = alloc(V0 * 32 * 2);
  size_t oTd0 = alloc(V1 * 32 * 2);
  size_t oT1a = alloc(V1 * 64 * 2);
  size_t oT1b = alloc(V1 * 64 * 2);
  size_t oTd1 = alloc(V2 * 128 * 2);
  size_t oT2a = alloc(V2 * 128 * 2);
  size_t oT2b = alloc(V2 * 128 * 2);
  size_t oT2c = alloc(V2 * 128 * 2);
  size_t actEnd = off;
  // f32 fields
  size_t oM0  = alloc(262144 * 4);
  size_t oM1  = alloc(32768 * 4);
  size_t oM2  = alloc(4096 * 4);
  size_t oPxy = alloc(32768 * 4);
  size_t oPyz = alloc(32768 * 4);
  size_t oPxz = alloc(32768 * 4);

  const int CinP_[9]  = {32, 32, 32, 32, 64, 64, 128, 128, 128};  // padded Cin
  const int srcC_[9]  = {16, 32, 32, 32, 64, 64, 128, 128, 128};  // real Cin
  const int Cout_[9]  = {32, 32, 32, 64, 64, 128, 128, 128, 128};
  const int Dout_[9]  = {64, 64, 32, 32, 32, 16, 16, 16, 16};
  const int strd_[9]  = {1, 1, 2, 1, 1, 2, 1, 1, 1};
  const int Pin_[9]   = {66, 66, 66, 34, 34, 34, 18, 18, 18};
  const int Pout_[9]  = {66, 66, 34, 34, 34, 18, 18, 18, 18};
  const size_t inOff_[9]  = {oTin, oT0a, oT0b, oTd0, oT1a, oT1b, oTd1, oT2a, oT2b};
  const size_t outOff_[9] = {oT0a, oT0b, oTd0, oT1a, oT1b, oTd1, oT2a, oT2b, oT2c};
  const int wIdx_[9]  = {3, 5, 7, 9, 11, 13, 15, 17, 19};
  const int bnIdx_[9] = {4, 6, 8, 10, 12, 14, 16, 18, 20};
  const size_t mOff_[9] = {oM0, oM0, oM1, oM1, oM1, oM2, oM2, oM2, oM2};

  size_t oScale[9], oShift[9], oPack[9];
  int nSteps[9];
  for (int l = 0; l < 9; ++l) {
    nSteps[l] = 27 * CinP_[l] / 32;
    oScale[l] = alloc(Cout_[l] * 4);
    oShift[l] = alloc(Cout_[l] * 4);
    oPack[l]  = alloc((size_t)32 * Cout_[l] * nSteps[l] * 2);
  }
  if (off > ws_size) return;  // workspace insufficient: nothing safe to do

  // 1) zero halo'd activation region (also provides the input channel padding)
  k_zero16<<<4096, 256, 0, stream>>>((uint4*)ws, actEnd / 16);

  // 2) masks at the three resolutions
  k_mask_to_float<<<1024, 256, 0, stream>>>(maskb, (float*)(ws + oM0), 262144);
  k_downsample_mask<<<128, 256, 0, stream>>>((const float*)(ws + oM0), (float*)(ws + oM1), 32, 64);
  k_downsample_mask<<<16, 256, 0, stream>>>((const float*)(ws + oM1), (float*)(ws + oM2), 16, 32);

  // 3) masked input -> f16 padded channels-last (32-ch stride)
  k_embed_input<<<16384, 256, 0, stream>>>(xin, (const float*)(ws + oM0), (_Float16*)(ws + oTin));

  // 4) per-layer BN folding + weight packing
  for (int l = 0; l < 9; ++l) {
    k_bn_prep<<<(Cout_[l] + 63) / 64, 64, 0, stream>>>(
        (const float*)d_in[bnIdx_[l]], (float*)(ws + oScale[l]), (float*)(ws + oShift[l]), Cout_[l]);
    int total = 32 * Cout_[l] * nSteps[l];
    k_pack_weights<<<(total + 255) / 256, 256, 0, stream>>>(
        (const float*)d_in[wIdx_[l]], (_Float16*)(ws + oPack[l]),
        CinP_[l], srcC_[l], Cout_[l], nSteps[l]);
  }

  // 5) nine fused implicit-GEMM WMMA conv layers (templated on <CIN, PIN>)
  for (int l = 0; l < 9; ++l) {
    int D = Dout_[l];
    dim3 grid((unsigned)(D * D * D / 16), (unsigned)(Cout_[l] / 32));
    const _Float16* inp = (const _Float16*)(ws + inOff_[l]);
    const _Float16* apk = (const _Float16*)(ws + oPack[l]);
    const float* sc = (const float*)(ws + oScale[l]);
    const float* sh = (const float*)(ws + oShift[l]);
    const float* mk = (const float*)(ws + mOff_[l]);
    _Float16* op = (_Float16*)(ws + outOff_[l]);
    int Po = Pout_[l], Co = Cout_[l], ds = lg2i(D), st = strd_[l];
    if (CinP_[l] == 32 && Pin_[l] == 66)
      k_conv3d_wmma<32, 66><<<grid, 32, 0, stream>>>(inp, apk, sc, sh, mk, op, Po, Co, D, ds, st);
    else if (CinP_[l] == 32 && Pin_[l] == 34)
      k_conv3d_wmma<32, 34><<<grid, 32, 0, stream>>>(inp, apk, sc, sh, mk, op, Po, Co, D, ds, st);
    else if (CinP_[l] == 64)
      k_conv3d_wmma<64, 34><<<grid, 32, 0, stream>>>(inp, apk, sc, sh, mk, op, Po, Co, D, ds, st);
    else
      k_conv3d_wmma<128, 18><<<grid, 32, 0, stream>>>(inp, apk, sc, sh, mk, op, Po, Co, D, ds, st);
  }

  // 6) triplane means + bilinear sampling
  k_make_planes<<<128, 256, 0, stream>>>((const _Float16*)(ws + oT2c),
                                         (float*)(ws + oPxy), (float*)(ws + oPyz),
                                         (float*)(ws + oPxz));
  k_sample<<<65536, 128, 0, stream>>>(crd, (const float*)(ws + oPxy),
                                      (const float*)(ws + oPyz), (const float*)(ws + oPxz),
                                      (float*)d_out);
}
